// RSAComponentModel_43361989820619
// MI455X (gfx1250) — matrix operand order repaired and checked by reading, not compiled, not run on hardware
//
#include <hip/hip_runtime.h>
#include <hip/hip_bf16.h>
#include <stdint.h>

// ---------------------------------------------------------------------------
// MI455X (gfx1250): bf16 WMMA (16x16x32, f32 accum), wave32.
//   convert: x -> bf16 [1024,224]; W0^T [2048,224]; W1^T [1024,2048];
//            HW1^T [128][512,1024]; HW2^T [128][256,512]
//   gemm0/1: trunk with BN+ReLU epilogues (bf16 out)
//   heads:   fused z=relu(feats@HW1+Hb1) (LDS bf16, A staged via async
//            global->LDS double buffer) then out=z@HW2+Hb2 (f32)
// ---------------------------------------------------------------------------

#define AS1 __attribute__((address_space(1)))
#define AS3 __attribute__((address_space(3)))

typedef __attribute__((ext_vector_type(16))) __bf16 bf16x16;
typedef __attribute__((ext_vector_type(8)))  float  f32x8;
typedef int v4i __attribute__((vector_size(16)));   // matches builtin prototype

union FragCast {
    struct { uint4 lo; uint4 hi; } u;
    bf16x16 v;
};

__device__ __forceinline__ uint16_t f32_to_bf16_bits(float x) {
    uint32_t u = __builtin_bit_cast(uint32_t, x);
    u += 0x7FFFu + ((u >> 16) & 1u);        // round-to-nearest-even
    return (uint16_t)(u >> 16);
}

// A fragment: 16x32 bf16 tile of row-major A (lda in elements).
__device__ __forceinline__ bf16x16 load_a_frag(const uint16_t* __restrict__ A,
                                               int lda, int row0, int k0, int lane) {
    int m = lane & 15, hi = lane >> 4;
    const uint16_t* p = A + (size_t)(row0 + m) * lda + k0 + hi * 8;
    FragCast f;
    f.u.lo = *(const uint4*)p;          // K = hi*8 .. hi*8+7
    f.u.hi = *(const uint4*)(p + 16);   // K = 16+hi*8 .. 16+hi*8+7
    return f.v;
}

// B fragment: 32x16 tile of B stored as B^T row-major [N,K].
__device__ __forceinline__ bf16x16 load_b_frag(const uint16_t* __restrict__ BT,
                                               int ldb, int col0, int k0, int lane) {
    int n = lane & 15, hi = lane >> 4;
    const uint16_t* p = BT + (size_t)(col0 + n) * ldb + k0 + hi * 8;
    FragCast f;
    f.u.lo = *(const uint4*)p;          // K = hi*8 .. hi*8+7
    f.u.hi = *(const uint4*)(p + 16);   // K = 16+hi*8 .. 16+hi*8+7
    return f.v;
}

__device__ __forceinline__ f32x8 wmma_bf16(bf16x16 a, bf16x16 b, f32x8 c) {
    return __builtin_amdgcn_wmma_f32_16x16x32_bf16(
        false, a, false, b, (short)0, c, false, false);
}

// ---- async global->LDS (ASYNCcnt-tracked), builtin if present else asm ----
__device__ __forceinline__ void async_load_b128(const uint16_t* g, uint16_t* l) {
#if __has_builtin(__builtin_amdgcn_global_load_async_to_lds_b128)
    __builtin_amdgcn_global_load_async_to_lds_b128(
        (AS1 v4i*)g, (AS3 v4i*)l, 0, 0);
#else
    asm volatile("global_load_async_to_lds_b128 %0, %1, off"
                 :: "v"((uint32_t)(size_t)(AS3 void*)l), "v"(g)
                 : "memory");
#endif
}

__device__ __forceinline__ void async_wait0() {
#if __has_builtin(__builtin_amdgcn_s_wait_asynccnt)
    __builtin_amdgcn_s_wait_asynccnt(0);
#else
    asm volatile("s_wait_asynccnt 0x0" ::: "memory");
#endif
}

// ---------------------------------------------------------------------------
// Conversion kernels
// ---------------------------------------------------------------------------
__global__ void k_pad_bf16(const float* __restrict__ in, uint16_t* __restrict__ out,
                           int R, int Kin, int Kp) {
    int tid = blockIdx.x * blockDim.x + threadIdx.x;
    if (tid >= R * Kp) return;
    int r = tid / Kp, k = tid - r * Kp;
    float v = (k < Kin) ? in[(size_t)r * Kin + k] : 0.0f;
    out[tid] = f32_to_bf16_bits(v);
}

// in [K,N] f32 (batched) -> out [N,Kp] bf16 (transposed, zero-padded K).
__global__ void k_transpose_bf16(const float* __restrict__ in, uint16_t* __restrict__ out,
                                 int K, int N, int Kp,
                                 long inBatchStride, long outBatchStride) {
    __shared__ float tile[32][33];
    const float* inb = in + (size_t)blockIdx.z * inBatchStride;
    uint16_t* outb   = out + (size_t)blockIdx.z * outBatchStride;
    int k0 = blockIdx.x * 32, n0 = blockIdx.y * 32;
    int tx = threadIdx.x & 31, ty = threadIdx.x >> 5;
#pragma unroll
    for (int i = 0; i < 4; ++i) {
        int k = k0 + ty + i * 8, n = n0 + tx;
        float v = (k < K && n < N) ? inb[(size_t)k * N + n] : 0.0f;
        tile[ty + i * 8][tx] = v;
    }
    __syncthreads();
#pragma unroll
    for (int i = 0; i < 4; ++i) {
        int n = n0 + ty + i * 8, k = k0 + tx;
        if (n < N && k < Kp)
            outb[(size_t)n * Kp + k] = f32_to_bf16_bits(tile[tx][ty + i * 8]);
    }
}

// ---------------------------------------------------------------------------
// Trunk GEMM: Out = relu(bn(A @ B^T + bias)), bf16 in/out, f32 accum.
// block = 256 (8 waves), block tile 128x64, wave tile 32x32 (2x2 WMMA).
// ---------------------------------------------------------------------------
__global__ void k_gemm_bn_relu(const uint16_t* __restrict__ A,
                               const uint16_t* __restrict__ BT,
                               uint16_t* __restrict__ Out,
                               int M, int N, int K,
                               const float* __restrict__ bias,
                               const float* __restrict__ g,
                               const float* __restrict__ be,
                               const float* __restrict__ mean,
                               const float* __restrict__ var) {
    int lane = threadIdx.x & 31;
    int w    = threadIdx.x >> 5;
    int wm   = w & 3, wn = w >> 2;
    int row0 = blockIdx.y * 128 + wm * 32;
    int col0 = blockIdx.x * 64  + wn * 32;

    f32x8 acc[2][2] = {};
    for (int k0 = 0; k0 < K; k0 += 32) {
        bf16x16 a0 = load_a_frag(A, K, row0,      k0, lane);
        bf16x16 a1 = load_a_frag(A, K, row0 + 16, k0, lane);
        bf16x16 b0 = load_b_frag(BT, K, col0,      k0, lane);
        bf16x16 b1 = load_b_frag(BT, K, col0 + 16, k0, lane);
        acc[0][0] = wmma_bf16(a0, b0, acc[0][0]);
        acc[0][1] = wmma_bf16(a0, b1, acc[0][1]);
        acc[1][0] = wmma_bf16(a1, b0, acc[1][0]);
        acc[1][1] = wmma_bf16(a1, b1, acc[1][1]);
    }

    int n = lane & 15, hi = lane >> 4;
#pragma unroll
    for (int tj = 0; tj < 2; ++tj) {
        int col = col0 + tj * 16 + n;
        float s = g[col] * rsqrtf(var[col] + 1e-5f);
        float t = be[col] - mean[col] * s;
        float b = bias[col];
#pragma unroll
        for (int ti = 0; ti < 2; ++ti) {
#pragma unroll
            for (int r = 0; r < 8; ++r) {
                int row = row0 + ti * 16 + hi * 8 + r;
                float y = (acc[ti][tj][r] + b) * s + t;
                Out[(size_t)row * N + col] = f32_to_bf16_bits(fmaxf(y, 0.0f));
            }
        }
    }
}

// ---------------------------------------------------------------------------
// Fused per-head kernel. blockIdx.x = head (128), blockIdx.y = 64-row slice (16).
// LDS: sz [64][512] bf16 (64 KB) + sA double buffer 2x[64][128] bf16 (32 KB).
// ---------------------------------------------------------------------------
__global__ void k_heads_fused(const uint16_t* __restrict__ F,     // [1024,1024] bf16
                              const uint16_t* __restrict__ HW1T,  // [128][512,1024]
                              const float* __restrict__ Hb1,      // [128,512]
                              const uint16_t* __restrict__ HW2T,  // [128][256,512]
                              const float* __restrict__ Hb2,      // [128,256]
                              float* __restrict__ Out) {          // [1024,128,256] f32
    extern __shared__ __align__(16) uint16_t smem[];
    uint16_t* sz = smem;                   // 64*512
    uint16_t* sA = smem + 64 * 512;        // 2 * 64*128
    const int head = blockIdx.x;
    const int rowBase = blockIdx.y * 64;
    const int t    = threadIdx.x;
    const int lane = t & 31;
    const int w    = t >> 5;

    const uint16_t* B1 = HW1T + (size_t)head * 512 * 1024;
    const uint16_t* B2 = HW2T + (size_t)head * 256 * 512;
    const float* hb1 = Hb1 + head * 512;
    const float* hb2 = Hb2 + head * 256;
    const int n = lane & 15, hi = lane >> 4;

    // stage a 64x128 bf16 chunk (16 KB): 4 async b128 per thread, contiguous 64B
    auto issue_chunk = [&](int c, int buf) {
#pragma unroll
        for (int i = 0; i < 4; ++i) {
            int vi   = t * 4 + i;            // 0..1023 16-byte vectors
            int row  = vi >> 4;              // 16 vectors per 128-elem row
            int koff = (vi & 15) * 8;
            const uint16_t* gsrc = F + (size_t)(rowBase + row) * 1024 + c * 128 + koff;
            uint16_t* ldst = sA + buf * (64 * 128) + row * 128 + koff;
            async_load_b128(gsrc, ldst);
        }
    };

    // ---- Stage 1: z[64,512] = relu(feats @ HW1^T + Hb1) -> bf16 in sz ----
    for (int cg = 0; cg < 2; ++cg) {                 // column-group: 256 cols each
        f32x8 acc[4][2] = {};
        issue_chunk(0, 0);
        for (int c = 0; c < 8; ++c) {                // 8 K-chunks of 128
            async_wait0();                           // my chunk-c loads landed
            __syncthreads();                         // whole tile visible; safe reuse
            if (c < 7) issue_chunk(c + 1, (c + 1) & 1);
            const uint16_t* Abuf = sA + (c & 1) * (64 * 128);
            for (int kk = 0; kk < 128; kk += 32) {
                int gk = c * 128 + kk;
                bf16x16 bfr[2];
#pragma unroll
                for (int tj = 0; tj < 2; ++tj)
                    bfr[tj] = load_b_frag(B1, 1024, cg * 256 + w * 32 + tj * 16, gk, lane);
#pragma unroll
                for (int ti = 0; ti < 4; ++ti) {
                    bf16x16 a = load_a_frag(Abuf, 128, ti * 16, kk, lane);
                    acc[ti][0] = wmma_bf16(a, bfr[0], acc[ti][0]);
                    acc[ti][1] = wmma_bf16(a, bfr[1], acc[ti][1]);
                }
            }
        }
        // bias + relu -> bf16 into sz (cols owned exclusively by this wave)
#pragma unroll
        for (int tj = 0; tj < 2; ++tj) {
            int col = cg * 256 + w * 32 + tj * 16 + n;
            float b = hb1[col];
#pragma unroll
            for (int ti = 0; ti < 4; ++ti) {
#pragma unroll
                for (int r = 0; r < 8; ++r) {
                    int rowl = ti * 16 + hi * 8 + r;
                    sz[rowl * 512 + col] = f32_to_bf16_bits(fmaxf(acc[ti][tj][r] + b, 0.0f));
                }
            }
        }
    }
    __syncthreads();

    // ---- Stage 2: out[64,256] = z @ HW2^T + Hb2; wave tile 32x64 ----
    {
        const int row0l = (w & 1) * 32;
        const int col0  = (w >> 1) * 64;
        f32x8 acc[2][4] = {};
        for (int k0 = 0; k0 < 512; k0 += 32) {
            bf16x16 a0 = load_a_frag(sz, 512, row0l,      k0, lane);
            bf16x16 a1 = load_a_frag(sz, 512, row0l + 16, k0, lane);
            bf16x16 bfr[4];
#pragma unroll
            for (int j = 0; j < 4; ++j)
                bfr[j] = load_b_frag(B2, 512, col0 + j * 16, k0, lane);
#pragma unroll
            for (int j = 0; j < 4; ++j) {
                acc[0][j] = wmma_bf16(a0, bfr[j], acc[0][j]);
                acc[1][j] = wmma_bf16(a1, bfr[j], acc[1][j]);
            }
        }
#pragma unroll
        for (int j = 0; j < 4; ++j) {
            int col = col0 + j * 16 + n;
            float b = hb2[col];
#pragma unroll
            for (int ti = 0; ti < 2; ++ti) {
#pragma unroll
                for (int r = 0; r < 8; ++r) {
                    int row = rowBase + row0l + ti * 16 + hi * 8 + r;
                    Out[(size_t)row * (128 * 256) + (size_t)head * 256 + col] =
                        acc[ti][j][r] + b;
                }
            }
        }
    }
}

// ---------------------------------------------------------------------------
// Host-side launcher
// ---------------------------------------------------------------------------
extern "C" void kernel_launch(void* const* d_in, const int* in_sizes, int n_in,
                              void* d_out, int out_size, void* d_ws, size_t ws_size,
                              hipStream_t stream) {
    (void)in_sizes; (void)n_in; (void)out_size; (void)ws_size;

    const float* x   = (const float*)d_in[0];
    const float* W0  = (const float*)d_in[1];
    const float* b0  = (const float*)d_in[2];
    const float* g0  = (const float*)d_in[3];
    const float* be0 = (const float*)d_in[4];
    const float* m0  = (const float*)d_in[5];
    const float* v0  = (const float*)d_in[6];
    const float* W1  = (const float*)d_in[7];
    const float* b1  = (const float*)d_in[8];
    const float* g1  = (const float*)d_in[9];
    const float* be1 = (const float*)d_in[10];
    const float* m1  = (const float*)d_in[11];
    const float* v1  = (const float*)d_in[12];
    const float* HW1 = (const float*)d_in[13];
    const float* Hb1 = (const float*)d_in[14];
    const float* HW2 = (const float*)d_in[15];
    const float* Hb2 = (const float*)d_in[16];
    float* out = (float*)d_out;

    char* ws = (char*)d_ws;
    uint16_t* xb    = (uint16_t*)(ws + 0);          // 1024x224 bf16
    uint16_t* w0t   = (uint16_t*)(ws + 458752);     // 2048x224
    uint16_t* hbuf  = (uint16_t*)(ws + 1376256);    // 1024x2048
    uint16_t* w1t   = (uint16_t*)(ws + 5570560);    // 1024x2048
    uint16_t* feats = (uint16_t*)(ws + 9764864);    // 1024x1024
    uint16_t* hw1t  = (uint16_t*)(ws + 11862016);   // 128x512x1024
    uint16_t* hw2t  = (uint16_t*)(ws + 146079744);  // 128x256x512

    // --- conversions ---
    k_pad_bf16<<<dim3((1024 * 224) / 256), dim3(256), 0, stream>>>(x, xb, 1024, 200, 224);
    k_transpose_bf16<<<dim3(7, 64, 1),    dim3(256), 0, stream>>>(W0, w0t, 200, 2048, 224, 0, 0);
    k_transpose_bf16<<<dim3(64, 32, 1),   dim3(256), 0, stream>>>(W1, w1t, 2048, 1024, 2048, 0, 0);
    k_transpose_bf16<<<dim3(32, 16, 128), dim3(256), 0, stream>>>(HW1, hw1t, 1024, 512, 1024,
                                                                  1024L * 512, 512L * 1024);
    k_transpose_bf16<<<dim3(16, 8, 128),  dim3(256), 0, stream>>>(HW2, hw2t, 512, 256, 512,
                                                                  512L * 256, 256L * 512);

    // --- trunk ---
    k_gemm_bn_relu<<<dim3(2048 / 64, 1024 / 128), dim3(256), 0, stream>>>(
        xb, w0t, hbuf, 1024, 2048, 224, b0, g0, be0, m0, v0);
    k_gemm_bn_relu<<<dim3(1024 / 64, 1024 / 128), dim3(256), 0, stream>>>(
        hbuf, w1t, feats, 1024, 1024, 2048, b1, g1, be1, m1, v1);

    // --- fused heads: 96 KB dynamic LDS (64 KB z + 2x16 KB A double buffer) ---
    const size_t heads_lds = (64 * 512 + 2 * 64 * 128) * sizeof(uint16_t);   // 98304
    (void)hipFuncSetAttribute((const void*)k_heads_fused,
                              hipFuncAttributeMaxDynamicSharedMemorySize,
                              (int)heads_lds);
    k_heads_fused<<<dim3(128, 16), dim3(256), heads_lds, stream>>>(
        feats, hw1t, Hb1, hw2t, Hb2, out);
}
